// LinearAttention_54219667145534
// MI455X (gfx1250) — matrix operand order repaired and checked by reading, not compiled, not run on hardware
//
#include <hip/hip_runtime.h>
#include <hip/hip_bf16.h>

// Linear attention: out = (q' @ (k'^T v)) / (q' @ sum_n k' + eps), x' = elu(x)+1
// B=4, N=4096, D=1024, fp32 in/out. WMMA f16 path with f32 accumulation.
// Block tile 128x256, 8 waves, each wave computes 4x4 WMMA tiles (64x64).

#define BATCH 4
#define NTOK  4096
#define DDIM  1024
#define EPSF  1e-6f

#define BM 128      // output tile rows per block
#define BN 256      // output tile cols per block
#define KC 64       // K chunk staged in LDS per iteration (2 WMMA k-steps)
#define KP 80       // padded K stride in halfs (16B-aligned, conflict-reducing)

typedef __attribute__((ext_vector_type(16))) _Float16 v16h;
typedef __attribute__((ext_vector_type(8)))  _Float16 v8h;
typedef __attribute__((ext_vector_type(4)))  _Float16 v4h;
typedef __attribute__((ext_vector_type(8)))  float    v8f;
typedef __attribute__((ext_vector_type(4)))  float    v4f;

__device__ __forceinline__ float elu1(float x) {
    return x > 0.0f ? x + 1.0f : __expf(x);
}

union FragU { v16h v; v8h h[2]; };

// ---------------------------------------------------------------------------
// Kernel 1: ksum[b][d] = sum_n elu1(K[b][n][d])   (coalesced column sums)
// ---------------------------------------------------------------------------
__global__ void __launch_bounds__(256) ksum_kernel(const float* __restrict__ K,
                                                   float* __restrict__ ksum) {
    int idx = blockIdx.x * 256 + threadIdx.x;       // 0 .. B*D-1
    int b = idx / DDIM;
    int d = idx % DDIM;
    const float* p = K + (size_t)b * NTOK * DDIM + d;
    float s = 0.0f;
    for (int n = 0; n < NTOK; ++n)
        s += elu1(p[(size_t)n * DDIM]);
    ksum[idx] = s;
}

// ---------------------------------------------------------------------------
// Kernel 2: den[b][n] = sum_d elu1(Q[b][n][d]) * ksum[b][d]  (one wave per row)
// ---------------------------------------------------------------------------
__global__ void __launch_bounds__(256) den_kernel(const float* __restrict__ Q,
                                                  const float* __restrict__ ksum,
                                                  float* __restrict__ den) {
    int wave = threadIdx.x >> 5;
    int lane = threadIdx.x & 31;
    int row  = blockIdx.x * 8 + wave;               // 0 .. B*N-1
    int b = row / NTOK;
    const float* q = Q + (size_t)row * DDIM;
    const float* s = ksum + (size_t)b * DDIM;
    float acc = 0.0f;
    for (int d = lane; d < DDIM; d += 32)
        acc += elu1(q[d]) * s[d];
    #pragma unroll
    for (int off = 16; off > 0; off >>= 1)
        acc += __shfl_xor(acc, off, 32);
    if (lane == 0) den[row] = acc;
}

// ---------------------------------------------------------------------------
// Shared WMMA inner step: load 4 A-frags + 4 B-frags from LDS, 16 WMMAs.
// Fragment layouts per CDNA5 ISA:
//   A (16x32 f16): lane l16 = M, runs of 8 halfs at k = lh*8 and lh*8+16
//   B (32x16 f16): lane l16 = N, runs of 8 halfs at k = lh*8 and lh*8+16
//   C (16x16 f32): element j -> M = j + 8*lh, N = l16
// ---------------------------------------------------------------------------
__device__ __forceinline__ void wmma_step(const _Float16* A_lds, const _Float16* B_lds,
                                          int wm, int wn, int l16, int lh, int ko,
                                          v8f acc[4][4]) {
    v16h a[4], bb[4];
    #pragma unroll
    for (int ms = 0; ms < 4; ++ms) {
        int m = (wm * 4 + ms) * 16 + l16;
        const v8h* p = (const v8h*)&A_lds[m * KP + ko * 32 + lh * 8];
        FragU u; u.h[0] = p[0]; u.h[1] = p[2];   // K runs [h*8..], [h*8+16..]
        a[ms] = u.v;
    }
    #pragma unroll
    for (int ns = 0; ns < 4; ++ns) {
        int e = (wn * 4 + ns) * 16 + l16;
        const v8h* p = (const v8h*)&B_lds[e * KP + ko * 32 + lh * 8];
        FragU u; u.h[0] = p[0]; u.h[1] = p[2];   // K runs [h*8..], [h*8+16..]
        bb[ns] = u.v;
    }
    #pragma unroll
    for (int ms = 0; ms < 4; ++ms)
        #pragma unroll
        for (int ns = 0; ns < 4; ++ns)
            acc[ms][ns] = __builtin_amdgcn_wmma_f32_16x16x32_f16(
                false, a[ms], false, bb[ns], (short)0, acc[ms][ns],
                false, false);
}

// ---------------------------------------------------------------------------
// Kernel 3: kv[b] = k'^T v   ([D x D] = [D x N] * [N x D], K-dim = N)
// ---------------------------------------------------------------------------
__global__ void __launch_bounds__(256, 2) kv_gemm(const float* __restrict__ Kmat,
                                                  const float* __restrict__ V,
                                                  float* __restrict__ kv) {
    __shared__ alignas(16) _Float16 A_lds[BM * KP];   // [m(d-local)][k(n-local)]
    __shared__ alignas(16) _Float16 B_lds[BN * KP];   // [e-local][k(n-local)] (B^T)

    const int tid  = threadIdx.x;
    const int b    = blockIdx.z;
    const int d0   = blockIdx.x * BM;
    const int e0   = blockIdx.y * BN;
    const int wave = tid >> 5;
    const int lane = tid & 31;
    const int wm   = wave >> 2;      // 0..1 : group of 4 m-subtiles
    const int wn   = wave & 3;       // 0..3 : group of 4 n-subtiles
    const int lh   = lane >> 4;      // half index h
    const int l16  = lane & 15;

    v8f acc[4][4] = {};

    for (int n0 = 0; n0 < NTOK; n0 += KC) {
        // Stage A: A_lds[m*KP + k] = elu1(K[b][n0+k][d0+m])  (transpose in LDS)
        {
            const float* src = Kmat + ((size_t)b * NTOK + n0) * DDIM + d0;
            #pragma unroll
            for (int i = 0; i < 8; ++i) {
                int c  = tid + i * 256;      // 0..2047
                int kr = c >> 5;             // n-local 0..63
                int m4 = (c & 31) << 2;      // d-local 0,4,...,124
                v4f x = *(const v4f*)(src + (size_t)kr * DDIM + m4);
                A_lds[(m4 + 0) * KP + kr] = (_Float16)elu1(x.x);
                A_lds[(m4 + 1) * KP + kr] = (_Float16)elu1(x.y);
                A_lds[(m4 + 2) * KP + kr] = (_Float16)elu1(x.z);
                A_lds[(m4 + 3) * KP + kr] = (_Float16)elu1(x.w);
            }
            const float* srcv = V + ((size_t)b * NTOK + n0) * DDIM + e0;
            #pragma unroll
            for (int i = 0; i < 16; ++i) {
                int c  = tid + i * 256;      // 0..4095
                int kr = c >> 6;             // n-local 0..63
                int e4 = (c & 63) << 2;      // e-local 0,4,...,252
                v4f x = *(const v4f*)(srcv + (size_t)kr * DDIM + e4);
                B_lds[(e4 + 0) * KP + kr] = (_Float16)x.x;
                B_lds[(e4 + 1) * KP + kr] = (_Float16)x.y;
                B_lds[(e4 + 2) * KP + kr] = (_Float16)x.z;
                B_lds[(e4 + 3) * KP + kr] = (_Float16)x.w;
            }
        }
        __syncthreads();
        #pragma unroll
        for (int ko = 0; ko < 2; ++ko)
            wmma_step(A_lds, B_lds, wm, wn, l16, lh, ko, acc);
        __syncthreads();
    }

    // Writeback: C layout M = j + 8*lh, N = l16
    #pragma unroll
    for (int ms = 0; ms < 4; ++ms) {
        int mrow0 = d0 + (wm * 4 + ms) * 16 + lh * 8;
        #pragma unroll
        for (int ns = 0; ns < 4; ++ns) {
            int col = e0 + (wn * 4 + ns) * 16 + l16;
            float* dst = kv + ((size_t)b * DDIM + mrow0) * DDIM + col;
            #pragma unroll
            for (int j = 0; j < 8; ++j)
                dst[(size_t)j * DDIM] = acc[ms][ns][j];
        }
    }
}

// ---------------------------------------------------------------------------
// Kernel 4: out[b] = (q' @ kv[b]) / (den + eps)   (K-dim = D)
// ---------------------------------------------------------------------------
__global__ void __launch_bounds__(256, 2) out_gemm(const float* __restrict__ Q,
                                                   const float* __restrict__ kv,
                                                   const float* __restrict__ den,
                                                   float* __restrict__ out) {
    __shared__ alignas(16) _Float16 A_lds[BM * KP];   // [m(token)][k(d-local)]
    __shared__ alignas(16) _Float16 B_lds[BN * KP];   // [e-local][k(d-local)] (kv^T)

    const int tid  = threadIdx.x;
    const int b    = blockIdx.z;
    const int n0   = blockIdx.x * BM;    // token rows
    const int e0   = blockIdx.y * BN;
    const int wave = tid >> 5;
    const int lane = tid & 31;
    const int wm   = wave >> 2;
    const int wn   = wave & 3;
    const int lh   = lane >> 4;
    const int l16  = lane & 15;

    v8f acc[4][4] = {};

    for (int k0 = 0; k0 < DDIM; k0 += KC) {
        // Stage A (row-major, elu): A_lds[m*KP + k] = elu1(Q[b][n0+m][k0+k])
        // Contiguous along k -> pack 4 halfs, one ds_store_b64.
        {
            const float* src = Q + ((size_t)b * NTOK + n0) * DDIM + k0;
            #pragma unroll
            for (int i = 0; i < 8; ++i) {
                int c  = tid + i * 256;
                int m  = c >> 4;             // row 0..127
                int k4 = (c & 15) << 2;      // 0,4,...,60
                v4f x = *(const v4f*)(src + (size_t)m * DDIM + k4);
                v4h h;
                h[0] = (_Float16)elu1(x.x);
                h[1] = (_Float16)elu1(x.y);
                h[2] = (_Float16)elu1(x.z);
                h[3] = (_Float16)elu1(x.w);
                *(v4h*)&A_lds[m * KP + k4] = h;
            }
            // Stage B^T: B_lds[e*KP + k] = kv[b][k0+k][e0+e]  (transpose in LDS)
            const float* srcb = kv + ((size_t)b * DDIM + k0) * DDIM + e0;
            #pragma unroll
            for (int i = 0; i < 16; ++i) {
                int c  = tid + i * 256;      // 0..4095
                int kr = c >> 6;             // 0..63
                int e4 = (c & 63) << 2;      // 0,4,...,252
                v4f x = *(const v4f*)(srcb + (size_t)kr * DDIM + e4);
                B_lds[(e4 + 0) * KP + kr] = (_Float16)x.x;
                B_lds[(e4 + 1) * KP + kr] = (_Float16)x.y;
                B_lds[(e4 + 2) * KP + kr] = (_Float16)x.z;
                B_lds[(e4 + 3) * KP + kr] = (_Float16)x.w;
            }
        }
        __syncthreads();
        #pragma unroll
        for (int ko = 0; ko < 2; ++ko)
            wmma_step(A_lds, B_lds, wm, wn, l16, lh, ko, acc);
        __syncthreads();
    }

    // Fused divide by denominator and writeback
    #pragma unroll
    for (int ms = 0; ms < 4; ++ms) {
        int row0 = n0 + (wm * 4 + ms) * 16 + lh * 8;
        #pragma unroll
        for (int ns = 0; ns < 4; ++ns) {
            int col = e0 + (wn * 4 + ns) * 16 + l16;
            #pragma unroll
            for (int j = 0; j < 8; ++j) {
                int row = row0 + j;
                float dv = den[(size_t)b * NTOK + row] + EPSF;
                out[((size_t)b * NTOK + row) * DDIM + col] = acc[ms][ns][j] / dv;
            }
        }
    }
}

// ---------------------------------------------------------------------------
extern "C" void kernel_launch(void* const* d_in, const int* in_sizes, int n_in,
                              void* d_out, int out_size, void* d_ws, size_t ws_size,
                              hipStream_t stream) {
    const float* q = (const float*)d_in[0];
    const float* k = (const float*)d_in[1];
    const float* v = (const float*)d_in[2];
    float* out = (float*)d_out;

    float* kv   = (float*)d_ws;                              // [B][D][D]  16 MB
    float* ksum = kv + (size_t)BATCH * DDIM * DDIM;          // [B][D]
    float* den  = ksum + (size_t)BATCH * DDIM;               // [B][N]

    ksum_kernel<<<(BATCH * DDIM) / 256, 256, 0, stream>>>(k, ksum);
    den_kernel<<<(BATCH * NTOK) / 8, 256, 0, stream>>>(q, ksum, den);
    kv_gemm<<<dim3(DDIM / BM, DDIM / BN, BATCH), 256, 0, stream>>>(k, v, kv);
    out_gemm<<<dim3(NTOK / BM, DDIM / BN, BATCH), 256, 0, stream>>>(q, kv, den, out);
}
